// TorchSpexAlchemicalModel_10015863734959
// MI455X (gfx1250) — compile-verified, matrix operand order checked by reading, not run on hardware
//
#include <hip/hip_runtime.h>

// ---------------------------------------------------------------------------
// TorchSpex alchemical model, MI455X (gfx1250, wave32) bf16-WMMA implementation
//
// Inputs (d_in order):
//  0: features [100000, 2048] f32     4: W3 [4,64,64] f32
//  1: W_comb   [4,4]          f32     5: W4 [4,1,64]  f32
//  2: W1       [4,64,2048]    f32     6: species [100000] i32
//  3: W2       [4,64,64]      f32     7: structure_indices [100000] i32
// Output: energies [1000] f32
//
// v2: 32 atoms per block (512 threads, 16 wave32). Each wave owns one N-tile
// (pseudo*4 + h-tile) and BOTH 16-atom M-sub-tiles, so every B fragment read
// from L2 feeds two v_wmma_f32_16x16x32_bf16. Halves W1 L2 traffic vs v1.
// ---------------------------------------------------------------------------

typedef __attribute__((ext_vector_type(16))) __bf16 v16bf;
typedef __attribute__((ext_vector_type(8)))  __bf16 v8bf;
typedef __attribute__((ext_vector_type(4)))  __bf16 v4bf;
typedef __attribute__((ext_vector_type(8)))  float  v8f;

#define NATOMS   100000
#define NFEAT    2048
#define NSPEC    4
#define NPSEUDO  4
#define NHID     64
#define NSTRUCT  1000
#define MTILE    32            // atoms per block
#define NTHREADS 512           // 16 wave32

// K index inside a 16-bit WMMA fragment: lane-half h, element e (0..15)
__device__ __host__ inline int frag_k(int half, int e) {
    return (e < 8) ? (half * 8 + e) : (16 + half * 8 + (e - 8));
}

__device__ inline float silu_n(float x) {
    float s = 1.0f / (1.0f + __expf(-x));
    return x * s * 1.6786f;
}

__device__ inline v16bf load_afrag(const __bf16* row) {
    v8bf alo = *(const v8bf*)row;
    v8bf ahi = *(const v8bf*)(row + 16);
    return __builtin_shufflevector(alo, ahi,
        0, 1, 2, 3, 4, 5, 6, 7, 8, 9, 10, 11, 12, 13, 14, 15);
}

// ---------------------------------------------------------------------------
// Kernel 0: zero output; convert W1/W2/W3 to bf16 pre-swizzled into WMMA
// B-fragment order. Fragment (nt, kt): 32 lanes x 16 contiguous bf16 per lane.
//   lane l: column n = l&15, half = l>>4 ; element e -> K = kt*32 + frag_k(...)
// W1 swizzled index: (((nt)*64 + kt)*32 + l)*16 + e,  nt = a*4 + ht (16 tiles)
// W2/W3 swizzled   : (((nt)*2  + kt)*32 + l)*16 + e
// ---------------------------------------------------------------------------
__global__ __launch_bounds__(256) void spex_init(
    const float* __restrict__ W1, const float* __restrict__ W2,
    const float* __restrict__ W3,
    __bf16* __restrict__ W1s, __bf16* __restrict__ W2s,
    __bf16* __restrict__ W3s, float* __restrict__ out)
{
    const int t = blockIdx.x * 256 + threadIdx.x;

    if (t < NSTRUCT) out[t] = 0.0f;

    // W1: 16 n-tiles * 64 k-tiles * 32 lanes * 16 elems = 524288
    if (t < (NPSEUDO * 4) * 64 * 32 * 16) {
        int e  = t & 15;
        int l  = (t >> 4) & 31;
        int kt = (t >> 9) & 63;
        int ht = (t >> 15) & 3;
        int a  = t >> 17;
        int k  = frag_k(l >> 4, e);
        int h  = ht * 16 + (l & 15);
        W1s[t] = (__bf16)W1[((size_t)(a * NHID + h)) * NFEAT + kt * 32 + k];
    }

    // W2/W3: 16 n-tiles * 2 k-tiles * 32 * 16 = 16384 each
    if (t < (NPSEUDO * 4) * 2 * 32 * 16) {
        int e  = t & 15;
        int l  = (t >> 4) & 31;
        int kt = (t >> 9) & 1;
        int ht = (t >> 10) & 3;
        int a  = t >> 12;
        int k  = frag_k(l >> 4, e);
        int h  = ht * 16 + (l & 15);
        int src = (a * NHID + h) * NHID + kt * 32 + k;
        W2s[t] = (__bf16)W2[src];
        W3s[t] = (__bf16)W3[src];
    }
}

// ---------------------------------------------------------------------------
// Main kernel: one 512-thread block (16 wave32) per 32 atoms.
// Dynamic LDS layout (bytes):
//   [0      ..131072)  xbf  : 32 x 2048 bf16 normalized features (A operand)
//   [131072 ..147456)  hA   : 32 x 256 bf16 hidden buffer
//   [147456 ..163840)  hB   : 32 x 256 bf16 hidden buffer
//   [163840 ..163968)  mean : 32 f32
//   [163968 ..164096)  rstd : 32 f32
//   [164096 ..164608)  wv   : 32 x 4 f32  (species->pseudo weights)
//   [164608 ..165120)  ered : 32 x 4 f32  (per-pseudo energies)
// ---------------------------------------------------------------------------
#define SMEM_BYTES 165120

__global__ __launch_bounds__(NTHREADS) void spex_main(
    const float* __restrict__ features, const float* __restrict__ Wc,
    const float* __restrict__ W4, const int* __restrict__ species,
    const int* __restrict__ sidx,
    const __bf16* __restrict__ W1s, const __bf16* __restrict__ W2s,
    const __bf16* __restrict__ W3s, float* __restrict__ out)
{
    extern __shared__ char smem[];
    __bf16* xbf   = (__bf16*)(smem);
    __bf16* hA    = (__bf16*)(smem + 131072);
    __bf16* hB    = (__bf16*)(smem + 147456);
    float*  meanv = (float*) (smem + 163840);
    float*  rstdv = (float*) (smem + 163968);
    float*  wv    = (float*) (smem + 164096);
    float*  ered  = (float*) (smem + 164608);

    const int t     = threadIdx.x;
    const int atom0 = blockIdx.x * MTILE;
    const int r  = t >> 4;        // atom row 0..31 (16 threads per row)
    const int cg = t & 15;        // column group within the row

    // ---- Pass 1: load features (f32, coalesced float4), accumulate stats,
    //              stash raw bf16 tile in LDS -------------------------------
    float s = 0.0f, s2 = 0.0f;
    const float4* frow = (const float4*)(features + (size_t)(atom0 + r) * NFEAT);
    #pragma unroll 4
    for (int j = 0; j < 32; ++j) {
        float4 v = frow[cg + 16 * j];
        s  += v.x + v.y + v.z + v.w;
        s2 += v.x * v.x + v.y * v.y + v.z * v.z + v.w * v.w;
        v4bf p;
        p[0] = (__bf16)v.x; p[1] = (__bf16)v.y;
        p[2] = (__bf16)v.z; p[3] = (__bf16)v.w;
        *(v4bf*)(xbf + r * NFEAT + (cg + 16 * j) * 4) = p;
    }
    // reduce across the 16 threads of this row (each row lives in one wave32)
    #pragma unroll
    for (int m = 8; m >= 1; m >>= 1) {
        s  += __shfl_xor(s,  m, 16);
        s2 += __shfl_xor(s2, m, 16);
    }
    if (cg == 0) {
        float mean = s * (1.0f / NFEAT);
        float var  = s2 * (1.0f / NFEAT) - mean * mean;
        meanv[r] = mean;
        rstdv[r] = rsqrtf(var);
    }
    // species -> pseudo combination weights for this atom tile
    if (t < MTILE * NPSEUDO) {
        int m = t >> 2, ap = t & 3;
        wv[m * 4 + ap] = Wc[ap * NSPEC + species[atom0 + m]];
    }
    __syncthreads();

    // ---- Pass 2: in-place elementwise normalization of the bf16 tile ------
    {
        float mean = meanv[r], rstd = rstdv[r];
        #pragma unroll 4
        for (int j = 0; j < 32; ++j) {
            v4bf* p = (v4bf*)(xbf + r * NFEAT + (cg + 16 * j) * 4);
            v4bf q = *p;
            #pragma unroll
            for (int k = 0; k < 4; ++k)
                q[k] = (__bf16)(((float)q[k] - mean) * rstd);
            *p = q;
        }
    }
    __syncthreads();

    // ---- WMMA setup: wave w owns N-tile nt = w, both 16-atom M-sub-tiles --
    const int lane = t & 31;
    const int nt   = t >> 5;       // 0..15 : pseudo = nt>>2, h-tile = nt&3
    const int half = lane >> 4;
    const int n16  = lane & 15;
    const int ap   = nt >> 2;

    // ---- Layer 1: two [16x2048] A tiles x one [2048x16] B column ----------
    v8f acc0 = {}, acc1 = {};      // M rows 0..15 / 16..31
    #pragma unroll 4
    for (int kt = 0; kt < 64; ++kt) {
        const __bf16* a0 = xbf + (size_t)n16 * NFEAT + kt * 32 + half * 8;
        const __bf16* a1 = a0 + 16 * NFEAT;
        v16bf af0 = load_afrag(a0);
        v16bf af1 = load_afrag(a1);
        const __bf16* bp = W1s + ((size_t)(nt * 64 + kt) * 32 + lane) * 16;
        if (kt + 2 < 64)
            __builtin_prefetch(bp + 2 * 32 * 16, 0, 3);
        v16bf b = *(const v16bf*)bp;
        acc0 = __builtin_amdgcn_wmma_f32_16x16x32_bf16(
            false, af0, false, b, (short)0, acc0, false, false);
        acc1 = __builtin_amdgcn_wmma_f32_16x16x32_bf16(
            false, af1, false, b, (short)0, acc1, false, false);
    }
    {
        const float inv_f = 0.022097086912079608f;  // 1/sqrt(2048)
        #pragma unroll
        for (int v = 0; v < 8; ++v) {
            int m0 = v + half * 8;          // sub-tile row
            int m1 = m0 + 16;
            hA[m0 * 256 + nt * 16 + n16] =
                (__bf16)silu_n(acc0[v] * inv_f * wv[m0 * 4 + ap]);
            hA[m1 * 256 + nt * 16 + n16] =
                (__bf16)silu_n(acc1[v] * inv_f * wv[m1 * 4 + ap]);
        }
    }
    __syncthreads();

    // ---- Layer 2: hA -> hB (K = 64, 2 k-tiles) ----------------------------
    {
        v8f c0 = {}, c1 = {};
        #pragma unroll
        for (int kt = 0; kt < 2; ++kt) {
            const __bf16* a0 = hA + n16 * 256 + ap * 64 + kt * 32 + half * 8;
            v16bf af0 = load_afrag(a0);
            v16bf af1 = load_afrag(a0 + 16 * 256);
            v16bf b = *(const v16bf*)(W2s + ((size_t)(nt * 2 + kt) * 32 + lane) * 16);
            c0 = __builtin_amdgcn_wmma_f32_16x16x32_bf16(
                false, af0, false, b, (short)0, c0, false, false);
            c1 = __builtin_amdgcn_wmma_f32_16x16x32_bf16(
                false, af1, false, b, (short)0, c1, false, false);
        }
        #pragma unroll
        for (int v = 0; v < 8; ++v) {
            int m0 = v + half * 8, m1 = m0 + 16;
            hB[m0 * 256 + nt * 16 + n16] = (__bf16)silu_n(c0[v] * 0.125f);
            hB[m1 * 256 + nt * 16 + n16] = (__bf16)silu_n(c1[v] * 0.125f);
        }
    }
    __syncthreads();

    // ---- Layer 3: hB -> hA ------------------------------------------------
    {
        v8f c0 = {}, c1 = {};
        #pragma unroll
        for (int kt = 0; kt < 2; ++kt) {
            const __bf16* a0 = hB + n16 * 256 + ap * 64 + kt * 32 + half * 8;
            v16bf af0 = load_afrag(a0);
            v16bf af1 = load_afrag(a0 + 16 * 256);
            v16bf b = *(const v16bf*)(W3s + ((size_t)(nt * 2 + kt) * 32 + lane) * 16);
            c0 = __builtin_amdgcn_wmma_f32_16x16x32_bf16(
                false, af0, false, b, (short)0, c0, false, false);
            c1 = __builtin_amdgcn_wmma_f32_16x16x32_bf16(
                false, af1, false, b, (short)0, c1, false, false);
        }
        #pragma unroll
        for (int v = 0; v < 8; ++v) {
            int m0 = v + half * 8, m1 = m0 + 16;
            hA[m0 * 256 + nt * 16 + n16] = (__bf16)silu_n(c0[v] * 0.125f);
            hA[m1 * 256 + nt * 16 + n16] = (__bf16)silu_n(c1[v] * 0.125f);
        }
    }
    __syncthreads();

    // ---- Layer 4 head + pseudo-sum + segment sum --------------------------
    if (t < MTILE * NPSEUDO) {     // 128 threads: (atom, pseudo) pairs
        int m = t & 31, a4 = t >> 5;
        const float* w4 = W4 + a4 * NHID;
        float sdot = 0.0f;
        #pragma unroll 8
        for (int j = 0; j < NHID; ++j)
            sdot += (float)hA[m * 256 + a4 * 64 + j] * w4[j];
        ered[m * 4 + a4] = sdot * 0.125f;   // inv_h
    }
    __syncthreads();
    if (t < MTILE) {
        int m = t;
        float tot = ered[m * 4] + ered[m * 4 + 1] + ered[m * 4 + 2] + ered[m * 4 + 3];
        // / sqrt(N_PSEUDO)=2 and / AVG_N_ATOMS=100  ->  * 0.005
        atomicAdd(out + sidx[atom0 + m], tot * 0.005f);
    }
}

extern "C" void kernel_launch(void* const* d_in, const int* in_sizes, int n_in,
                              void* d_out, int out_size, void* d_ws, size_t ws_size,
                              hipStream_t stream) {
    (void)in_sizes; (void)n_in; (void)out_size; (void)ws_size;
    const float* features = (const float*)d_in[0];
    const float* Wc       = (const float*)d_in[1];
    const float* W1       = (const float*)d_in[2];
    const float* W2       = (const float*)d_in[3];
    const float* W3       = (const float*)d_in[4];
    const float* W4       = (const float*)d_in[5];
    const int*   species  = (const int*)d_in[6];
    const int*   sidx     = (const int*)d_in[7];
    float*       out      = (float*)d_out;

    // Workspace layout (bf16): W1s 524288 | W2s 16384 | W3s 16384  (~1.1 MB)
    __bf16* W1s = (__bf16*)d_ws;
    __bf16* W2s = (__bf16*)((char*)d_ws + 524288 * 2);
    __bf16* W3s = (__bf16*)((char*)d_ws + 524288 * 2 + 16384 * 2);

    // Kernel 0: zero out + weight conversion/swizzle (524288 threads)
    spex_init<<<2048, 256, 0, stream>>>(W1, W2, W3, W1s, W2s, W3s, out);

    // Kernel 1: 32 atoms per block, 3125 blocks, 16 wave32 per block
    spex_main<<<NATOMS / MTILE, NTHREADS, SMEM_BYTES, stream>>>(
        features, Wc, W4, species, sidx, W1s, W2s, W3s, out);
}
